// AttentiveGAT_44238163149260
// MI455X (gfx1250) — matrix operand order repaired and checked by reading, not compiled, not run on hardware
//
#include <hip/hip_runtime.h>

// ---------------- problem constants (match reference) ----------------
#define N_NODES 50000
#define E_EDGES 800000
#define ET_EDGES (E_EDGES + N_NODES)   // with self loops
#define F 128                          // F_IN == HC == 128
#define H_HEADS 4
#define C_DIM 32
#define G_GRAPHS 64
#define OUT_C 10
#define NEG_SLOPE 0.2f

typedef __attribute__((ext_vector_type(16))) _Float16 v16h;
typedef __attribute__((ext_vector_type(8)))  float    v8f;

// float atomic max via int/uint ordering trick (correct for mixed signs)
__device__ __forceinline__ void atomicMaxFloat(float* addr, float value) {
    if (value >= 0.0f) atomicMax((int*)addr, __float_as_int(value));
    else               atomicMin((unsigned int*)addr, __float_as_uint(value));
}

__device__ __forceinline__ void edge_sd(const int* __restrict__ ei, int e, int& s, int& d) {
    if (e < E_EDGES) { s = ei[e]; d = ei[E_EDGES + e]; }
    else             { s = e - E_EDGES; d = s; }       // self loop
}

// ---------------------------------------------------------------------
// WMMA GEMM: Hout[N,128] = X[N,128] @ Wm[128,128]   (f16 mul, f32 acc)
// block = 256 thr = 8 waves; each wave computes a 16x128 row-stripe.
// W is staged in LDS pre-swizzled into B-fragment layout:
//   frag (kstep, ntile, lane) -> 16 contiguous halves.
//   lanes 0-15 : col = ntile*16+lane,      K = kstep*32 + 0..15
//   lanes 16-31: col = ntile*16+(lane-16), K = kstep*32 + 16..31
// ---------------------------------------------------------------------
__global__ __launch_bounds__(256)
void gat_gemm128(const float* __restrict__ X, const float* __restrict__ Wm,
                 float* __restrict__ Hout, int nrows) {
    __shared__ __attribute__((aligned(32))) _Float16 ldsW[4 * 8 * 32 * 16]; // 32KB

    const int tid = threadIdx.x;
    // cooperative swizzled fill of W fragments (1024 frag-rows / 256 thr)
    for (int j = 0; j < 4; ++j) {
        int idx   = j * 256 + tid;          // 0..1023
        int kstep = idx >> 8;
        int rem   = idx & 255;
        int ntile = rem >> 5;
        int lane  = rem & 31;
        int col   = ntile * 16 + (lane & 15);
        int kbase = kstep * 32 + ((lane >> 4) * 16);
        _Float16* dstp = &ldsW[idx * 16];
#pragma unroll
        for (int s = 0; s < 16; ++s)
            dstp[s] = (_Float16)Wm[(kbase + s) * F + col];
    }
    __syncthreads();

    const int wave    = tid >> 5;
    const int lane    = tid & 31;
    const int m       = lane & 15;
    const int halfsel = lane >> 4;
    const int rowbase = blockIdx.x * 128 + wave * 16;
    int row  = rowbase + m;
    int lrow = row < nrows ? row : (nrows - 1);          // clamp: keep EXEC all-ones
    const float* xr = X + (size_t)lrow * F;

    v8f acc[8];
#pragma unroll
    for (int nt = 0; nt < 8; ++nt) acc[nt] = (v8f){0, 0, 0, 0, 0, 0, 0, 0};

#pragma unroll
    for (int kstep = 0; kstep < 4; ++kstep) {
        // A fragment: slots 0-15 = K0..K0+15
        int k0 = kstep * 32 + halfsel * 16;
        v16h a;
#pragma unroll
        for (int s = 0; s < 8; ++s) a[s]     = (_Float16)xr[k0 + s];
#pragma unroll
        for (int s = 0; s < 8; ++s) a[8 + s] = (_Float16)xr[k0 + 8 + s];

#pragma unroll
        for (int nt = 0; nt < 8; ++nt) {
            const v16h b = *(const v16h*)&ldsW[(((kstep * 8) + nt) * 32 + lane) * 16];
            acc[nt] = __builtin_amdgcn_wmma_f32_16x16x32_f16(
                false, a, false, b, (short)0, acc[nt], false, false);
        }
    }

    // D layout: VGPR r -> M = r (lanes 0-15) / 8+r (lanes 16-31); N = lane&15
    const int rout0 = rowbase + (halfsel ? 8 : 0);
#pragma unroll
    for (int nt = 0; nt < 8; ++nt) {
        int col = nt * 16 + m;
#pragma unroll
        for (int r = 0; r < 8; ++r) {
            int ro = rout0 + r;
            if (ro < nrows) Hout[(size_t)ro * F + col] = acc[nt][r];
        }
    }
}

// alpha_s / alpha_d projections: one thread per (node, head)
__global__ void alpha_kernel(const float* __restrict__ Hm,
                             const float* __restrict__ a_src,
                             const float* __restrict__ a_dst,
                             float* __restrict__ as, float* __restrict__ ad) {
    int i = blockIdx.x * blockDim.x + threadIdx.x;
    if (i >= N_NODES * H_HEADS) return;
    int node = i >> 2, hh = i & 3;
    const float* hp  = Hm + (size_t)node * F + hh * C_DIM;
    const float* asp = a_src + hh * C_DIM;
    const float* adp = a_dst + hh * C_DIM;
    float ss = 0.f, sd = 0.f;
#pragma unroll
    for (int c = 0; c < C_DIM; ++c) { float v = hp[c]; ss += v * asp[c]; sd += v * adp[c]; }
    as[i] = ss; ad[i] = sd;
}

__global__ void init_layer(float* __restrict__ accum, float* __restrict__ emax,
                           float* __restrict__ denom) {
    int i = blockIdx.x * blockDim.x + threadIdx.x;
    if (i < N_NODES * F) accum[i] = 0.f;
    if (i < N_NODES * H_HEADS) { emax[i] = -3.0e38f; denom[i] = 0.f; }
}

// pass 1: segment max of leaky-relu logits (thread per edge*head)
__global__ void edge_max(const int* __restrict__ ei, const float* __restrict__ as,
                         const float* __restrict__ ad, float* __restrict__ emax) {
    int t = blockIdx.x * blockDim.x + threadIdx.x;
    if (t >= ET_EDGES * H_HEADS) return;
    int e = t >> 2, hh = t & 3;
    int s, d; edge_sd(ei, e, s, d);
    float v = as[s * H_HEADS + hh] + ad[d * H_HEADS + hh];
    v = v > 0.f ? v : NEG_SLOPE * v;
    atomicMaxFloat(&emax[d * H_HEADS + hh], v);
}

// pass 2: segment sum of exp(e - emax)
__global__ void edge_sum(const int* __restrict__ ei, const float* __restrict__ as,
                         const float* __restrict__ ad, const float* __restrict__ emax,
                         float* __restrict__ denom) {
    int t = blockIdx.x * blockDim.x + threadIdx.x;
    if (t >= ET_EDGES * H_HEADS) return;
    int e = t >> 2, hh = t & 3;
    int s, d; edge_sd(ei, e, s, d);
    float v = as[s * H_HEADS + hh] + ad[d * H_HEADS + hh];
    v = v > 0.f ? v : NEG_SLOPE * v;
    float ex = __expf(v - emax[d * H_HEADS + hh]);
    atomicAdd(&denom[d * H_HEADS + hh], ex);
}

// pass 3: weighted message scatter; one wave (32 lanes) per edge,
// lane l covers channel l of all 4 heads -> 4 coalesced 128B segments
__global__ __launch_bounds__(256)
void edge_msg(const int* __restrict__ ei, const float* __restrict__ as,
              const float* __restrict__ ad, const float* __restrict__ emax,
              const float* __restrict__ denom, const float* __restrict__ Hm,
              float* __restrict__ accum) {
    int e = blockIdx.x * 8 + (threadIdx.x >> 5);
    if (e >= ET_EDGES) return;
    int lane = threadIdx.x & 31;
    int s, d; edge_sd(ei, e, s, d);
    const float4 asv = *(const float4*)&as[s * 4];
    const float4 adv = *(const float4*)&ad[d * 4];
    const float4 emv = *(const float4*)&emax[d * 4];
    const float4 dnv = *(const float4*)&denom[d * 4];
    float al[4];
    {
        float v;
        v = asv.x + adv.x; v = v > 0.f ? v : NEG_SLOPE * v; al[0] = __expf(v - emv.x) / dnv.x;
        v = asv.y + adv.y; v = v > 0.f ? v : NEG_SLOPE * v; al[1] = __expf(v - emv.y) / dnv.y;
        v = asv.z + adv.z; v = v > 0.f ? v : NEG_SLOPE * v; al[2] = __expf(v - emv.z) / dnv.z;
        v = asv.w + adv.w; v = v > 0.f ? v : NEG_SLOPE * v; al[3] = __expf(v - emv.w) / dnv.w;
    }
    const float* hs = Hm + (size_t)s * F;
    float* ap = accum + (size_t)d * F;
#pragma unroll
    for (int hh = 0; hh < 4; ++hh)
        atomicAdd(&ap[hh * 32 + lane], al[hh] * hs[hh * 32 + lane]);
}

__global__ void bias_relu(const float* __restrict__ accum, const float* __restrict__ bias,
                          float* __restrict__ xout) {
    int i = blockIdx.x * blockDim.x + threadIdx.x;
    if (i >= N_NODES * F) return;
    float v = accum[i] + bias[i & 127];
    xout[i] = v > 0.f ? v : 0.f;
}

__global__ void pool_init(float* __restrict__ pooled) {
    int i = blockIdx.x * blockDim.x + threadIdx.x;
    if (i < G_GRAPHS * F) pooled[i] = 0.f;
}

__global__ void pool_kernel(const float* __restrict__ x, const int* __restrict__ batch,
                            float* __restrict__ pooled) {
    int i = blockIdx.x * blockDim.x + threadIdx.x;
    if (i >= N_NODES * F) return;
    int node = i >> 7, f = i & 127;
    atomicAdd(&pooled[batch[node] * F + f], x[i]);
}

__global__ void logits_kernel(const float* __restrict__ pooled, const float* __restrict__ Wh,
                              const float* __restrict__ bh, float* __restrict__ out) {
    int i = blockIdx.x * blockDim.x + threadIdx.x;
    if (i >= G_GRAPHS * OUT_C) return;
    int g = i / OUT_C, o = i % OUT_C;
    float sum = bh[o];
#pragma unroll 8
    for (int k = 0; k < F; ++k) sum += pooled[g * F + k] * Wh[k * OUT_C + o];
    out[i] = sum;
}

// ---------------------------------------------------------------------
extern "C" void kernel_launch(void* const* d_in, const int* in_sizes, int n_in,
                              void* d_out, int out_size, void* d_ws, size_t ws_size,
                              hipStream_t stream) {
    (void)in_sizes; (void)n_in; (void)out_size; (void)ws_size;

    const float* x        = (const float*)d_in[0];
    const float* Ws[3]    = {(const float*)d_in[1], (const float*)d_in[2], (const float*)d_in[3]};
    const float* a_srcs[3]= {(const float*)d_in[4], (const float*)d_in[5], (const float*)d_in[6]};
    const float* a_dsts[3]= {(const float*)d_in[7], (const float*)d_in[8], (const float*)d_in[9]};
    const float* biases[3]= {(const float*)d_in[10],(const float*)d_in[11],(const float*)d_in[12]};
    const float* Wh       = (const float*)d_in[13];
    const float* bh       = (const float*)d_in[14];
    const int*   ei       = (const int*)d_in[15];
    const int*   batch    = (const int*)d_in[16];

    // workspace layout (floats)
    float* ws    = (float*)d_ws;
    float* h     = ws;                                     // N*F
    float* accum = h     + (size_t)N_NODES * F;            // N*F
    float* xbuf  = accum + (size_t)N_NODES * F;            // N*F
    float* as    = xbuf  + (size_t)N_NODES * F;            // N*H
    float* ad    = as    + (size_t)N_NODES * H_HEADS;      // N*H
    float* emax  = ad    + (size_t)N_NODES * H_HEADS;      // N*H
    float* denom = emax  + (size_t)N_NODES * H_HEADS;      // N*H
    float* pooled= denom + (size_t)N_NODES * H_HEADS;      // G*F

    auto cdiv = [](int a, int b) { return (a + b - 1) / b; };
    const int gNF  = cdiv(N_NODES * F, 256);
    const int gNH  = cdiv(N_NODES * H_HEADS, 256);
    const int gEH  = cdiv(ET_EDGES * H_HEADS, 256);
    const int gE8  = cdiv(ET_EDGES, 8);
    const int gRow = cdiv(N_NODES, 128);

    const float* xin = x;
    for (int l = 0; l < 3; ++l) {
        gat_gemm128<<<gRow, 256, 0, stream>>>(xin, Ws[l], h, N_NODES);
        alpha_kernel<<<gNH, 256, 0, stream>>>(h, a_srcs[l], a_dsts[l], as, ad);
        init_layer<<<gNF, 256, 0, stream>>>(accum, emax, denom);
        edge_max<<<gEH, 256, 0, stream>>>(ei, as, ad, emax);
        edge_sum<<<gEH, 256, 0, stream>>>(ei, as, ad, emax, denom);
        edge_msg<<<gE8, 256, 0, stream>>>(ei, as, ad, emax, denom, h, accum);
        bias_relu<<<gNF, 256, 0, stream>>>(accum, biases[l], xbuf);
        xin = xbuf;
    }
    pool_init<<<cdiv(G_GRAPHS * F, 256), 256, 0, stream>>>(pooled);
    pool_kernel<<<gNF, 256, 0, stream>>>(xbuf, batch, pooled);
    logits_kernel<<<cdiv(G_GRAPHS * OUT_C, 256), 256, 0, stream>>>(pooled, Wh, bh, (float*)d_out);
}
